// Transformer_72430328480005
// MI455X (gfx1250) — matrix operand order repaired and checked by reading, not compiled, not run on hardware
//
#include <hip/hip_runtime.h>

// ---------------- model dims ----------------
#define BATCH  4
#define SEQL   1024
#define DMODEL 1024
#define NHEAD  16
#define DHEAD  64
#define DMLP   4096
#define DVOCAB 32000

typedef __attribute__((ext_vector_type(16))) __bf16 v16bf;
typedef __attribute__((ext_vector_type(8)))  float  v8f;
typedef __attribute__((ext_vector_type(4)))  unsigned int u32x4;
typedef __attribute__((ext_vector_type(8)))  int          i32x8;
typedef __attribute__((ext_vector_type(4)))  int          i32x4;

union Frag { v16bf v; unsigned int u[8]; };

#if __has_builtin(__builtin_amdgcn_tensor_load_to_lds) && __has_builtin(__builtin_amdgcn_s_wait_tensorcnt)
#define HAVE_TDM 1
#else
#define HAVE_TDM 0
#endif

__device__ __forceinline__ unsigned short f32_to_bf16(float f) {
  unsigned int u = __float_as_uint(f);
  u += 0x7FFFu + ((u >> 16) & 1u);     // round-to-nearest-even
  return (unsigned short)(u >> 16);
}

#if HAVE_TDM
// Issue a 2-D TDM tile load (bf16 elements): tile_d0 contiguous elems per row,
// tile_d1 rows, row stride = stride_elems. LDS dest is packed row-major.
__device__ __forceinline__ void tdm_load_2d(unsigned lds_off, const void* gptr,
                                            unsigned tile_d0, unsigned tile_d1,
                                            long long stride_elems) {
  unsigned long long ga = (unsigned long long)(size_t)gptr;
  u32x4 g0;
  g0[0] = 1u;                                         // count=1, user mode, no gather
  g0[1] = lds_off;                                    // LDS byte address
  g0[2] = (unsigned)(ga & 0xFFFFFFFFu);               // global_addr[31:0]
  g0[3] = (unsigned)((ga >> 32) & 0x1FFFFFFu) | (2u << 30);  // global_addr[56:32] | type=2
  unsigned long long st = (unsigned long long)stride_elems;
  i32x8 g1;
  g1[0] = (int)(1u << 16);                            // data_size=1 (2 bytes), mask=0
  g1[1] = (int)((tile_d0 & 0xFFFFu) << 16);           // tensor_dim0[15:0]  (== tile_d0)
  g1[2] = (int)((tile_d0 >> 16) | ((tile_d1 & 0xFFFFu) << 16));  // td0 hi | tensor_dim1 lo
  g1[3] = (int)((tile_d1 >> 16) | ((tile_d0 & 0xFFFFu) << 16));  // td1 hi | tile_dim0
  g1[4] = (int)(tile_d1 & 0xFFFFu);                   // tile_dim1 (tile_dim2 = 0)
  g1[5] = (int)(unsigned)(st & 0xFFFFFFFFu);          // tensor_dim0_stride[31:0]
  g1[6] = (int)(unsigned)((st >> 32) & 0xFFFFu);      // stride[47:32]; dim1_stride lo16 = 0
  g1[7] = 0;
  i32x4 z4 = {0, 0, 0, 0};
  i32x8 z8 = {0, 0, 0, 0, 0, 0, 0, 0};
  __builtin_amdgcn_tensor_load_to_lds(g0, g1, z4, z4, z8, 0);
}
#endif

// ---------------- f32 -> bf16 convert ----------------
__global__ void cvt_bf16_kernel(const float* __restrict__ in,
                                unsigned short* __restrict__ out, int n) {
  int i = blockIdx.x * 256 + threadIdx.x;
  if (i < n) out[i] = f32_to_bf16(in[i]);
}

// ---------------- embedding + positional ----------------
__global__ void embed_kernel(const int* __restrict__ x,
                             const float* __restrict__ emb,
                             const float* __restrict__ pos,
                             float* __restrict__ h,
                             unsigned short* __restrict__ hbf) {
  size_t i = (size_t)blockIdx.x * 256 + threadIdx.x;   // over B*S*D
  int d = (int)(i % DMODEL);
  size_t bs = i / DMODEL;
  int s = (int)(bs % SEQL);
  int tok = x[bs];
  float v = emb[(size_t)tok * DMODEL + d] + pos[(size_t)s * DMODEL + d];
  h[i] = v;
  hbf[i] = f32_to_bf16(v);
}

// ---------------- causal softmax (row-wise, in place f32 -> bf16) ----------------
__global__ __launch_bounds__(256) void softmax_causal_kernel(float* __restrict__ Sc) {
  int r = blockIdx.x;
  int srow = r % SEQL;
  float* rowp = Sc + (size_t)r * SEQL;
  int tid = threadIdx.x;

  float vals[4];
  float mx = -3.0e38f;
#pragma unroll
  for (int i = 0; i < 4; ++i) {
    int c = tid + i * 256;
    float v = (c <= srow) ? rowp[c] : -3.0e38f;
    vals[i] = v;
    mx = fmaxf(mx, v);
  }
  __shared__ float red[256];
  red[tid] = mx; __syncthreads();
  for (int s = 128; s > 0; s >>= 1) {
    if (tid < s) red[tid] = fmaxf(red[tid], red[tid + s]);
    __syncthreads();
  }
  mx = red[0]; __syncthreads();

  float sum = 0.0f;
#pragma unroll
  for (int i = 0; i < 4; ++i) {
    int c = tid + i * 256;
    float e = (c <= srow) ? __expf(vals[i] - mx) : 0.0f;
    vals[i] = e;
    sum += e;
  }
  red[tid] = sum; __syncthreads();
  for (int s = 128; s > 0; s >>= 1) {
    if (tid < s) red[tid] += red[tid + s];
    __syncthreads();
  }
  float inv = 1.0f / red[0];
  __syncthreads();                     // all row reads complete before overwrite

  unsigned short* P = (unsigned short*)rowp;   // bf16 probs, row stride 2*SEQL elems
#pragma unroll
  for (int i = 0; i < 4; ++i) {
    int c = tid + i * 256;
    P[c] = f32_to_bf16(vals[i] * inv);
  }
}

// ---------------- WMMA fragment compute for one 32-deep K step ----------------
template <int BN>
__device__ __forceinline__ void mma_step(const unsigned short* As_,
                                         const unsigned short* Bs_,
                                         v8f acc[][BN / 32],
                                         int wr, int wc, int half, int l15, int lane) {
  constexpr int TJ = BN / 32;
  Frag a[2], b[TJ];
#pragma unroll
  for (int ti = 0; ti < 2; ++ti) {
    int m = wr * 32 + ti * 16 + l15;
#pragma unroll
    for (int v = 0; v < 8; ++v) {
      int kk = ((v >> 2) << 4) + half * 8 + ((v & 3) << 1);
      a[ti].u[v] = *(const unsigned int*)(As_ + m * 32 + kk);
    }
  }
#pragma unroll
  for (int tj = 0; tj < TJ; ++tj) {
    int nb = wc * (BN / 2) + tj * 16;
    int n = nb + l15;
#pragma unroll
    for (int v = 0; v < 8; ++v) {
      int kk = ((v >> 2) << 4) + half * 8 + ((v & 3) << 1);
      b[tj].u[v] = (unsigned int)Bs_[kk * BN + n] | ((unsigned int)Bs_[(kk + 1) * BN + n] << 16);
    }
  }
#pragma unroll
  for (int ti = 0; ti < 2; ++ti)
#pragma unroll
    for (int tj = 0; tj < TJ; ++tj)
      acc[ti][tj] = __builtin_amdgcn_wmma_f32_16x16x32_bf16(
          false, a[ti].v, false, b[tj].v, (short)0, acc[ti][tj], false, false);
}

// ---------------- generic bf16 WMMA GEMM ----------------
// C[z] = alpha*A[z]*B[z] + bias + Res, optional ReLU, f32 and/or bf16 out.
// BM=128, BN template (64/128), BK=32. 8 waves: 4 row-bands x 2 col-bands.
// TRANSB=0: TDM double-buffered pipeline. TRANSB=1: vector-load transpose staging.
template <int BN, int TRANSB>
__global__ __launch_bounds__(256) void gemm_bf16_wmma(
    const unsigned short* __restrict__ A, const unsigned short* __restrict__ B,
    float* __restrict__ C, unsigned short* __restrict__ Cbf,
    const float* __restrict__ bias, const float* __restrict__ Res,
    int M, int N, int K, int lda, int ldb, int ldc,
    long long aDiv, long long aS1, long long aMod, long long aS2,
    long long bDiv, long long bS1, long long bMod, long long bS2,
    long long cDiv, long long cS1, long long cMod, long long cS2,
    int relu, float alpha) {
  constexpr int TJ = BN / 32;
  __shared__ __align__(16) unsigned short As[2][128 * 32];
  __shared__ __align__(16) unsigned short Bs[2][32 * BN];

  long long z = blockIdx.z;
  const unsigned short* Ab = A + (z / aDiv) * aS1 + (z % aMod) * aS2;
  const unsigned short* Bb = B + (z / bDiv) * bS1 + (z % bMod) * bS2;
  long long coff = (z / cDiv) * cS1 + (z % cMod) * cS2;

  int tid = threadIdx.x;
  int lane = tid & 31, w = tid >> 5;
  int wr = w & 3;            // 32-row band
  int wc = w >> 2;           // (BN/2)-col band
  int half = lane >> 4, l15 = lane & 15;
  int m0 = blockIdx.y * 128;
  int n0 = blockIdx.x * BN;

  v8f acc[2][TJ];
#pragma unroll
  for (int i = 0; i < 2; ++i)
#pragma unroll
    for (int j = 0; j < TJ; ++j)
#pragma unroll
      for (int v = 0; v < 8; ++v) acc[i][j][v] = 0.0f;

  int nsteps = K >> 5;

#if HAVE_TDM
  if constexpr (!TRANSB) {
    // ---- TDM double-buffered pipeline: wave 0 drives the Tensor Data Mover ----
    if (w == 0) {
      tdm_load_2d((unsigned)(size_t)&As[0][0], Ab + (long long)m0 * lda, 32, 128, lda);
      tdm_load_2d((unsigned)(size_t)&Bs[0][0], Bb + n0, BN, 32, ldb);
    }
    for (int i = 0; i < nsteps; ++i) {
      int cur = i & 1;
      if (i + 1 < nsteps) {
        if (w == 0) {
          long long k0n = (long long)(i + 1) << 5;
          tdm_load_2d((unsigned)(size_t)&As[cur ^ 1][0],
                      Ab + (long long)m0 * lda + k0n, 32, 128, lda);
          tdm_load_2d((unsigned)(size_t)&Bs[cur ^ 1][0],
                      Bb + k0n * ldb + n0, BN, 32, ldb);
        }
        __builtin_amdgcn_s_wait_tensorcnt((short)2);   // current tile retired
      } else {
        __builtin_amdgcn_s_wait_tensorcnt((short)0);
      }
      __syncthreads();
      mma_step<BN>(&As[cur][0], &Bs[cur][0], acc, wr, wc, half, l15, lane);
      __syncthreads();
    }
  } else
#endif
  {
    // ---- vector-load staging (transB, or no-TDM fallback) ----
    for (int i = 0; i < nsteps; ++i) {
      int k0 = i << 5;
      {  // A tile: 128x32, 16 bf16 per thread
        int row = tid >> 1;
        int col = (tid & 1) << 4;
        const uint4* g = (const uint4*)(Ab + (long long)(m0 + row) * lda + (k0 + col));
        uint4 u0 = g[0];
        uint4 u1 = g[1];
        uint4* s = (uint4*)(&As[0][0] + row * 32 + col);
        s[0] = u0; s[1] = u1;
      }
      if constexpr (!TRANSB) {
        // B tile 32xBN row-major
        if constexpr (BN == 128) {
          int row = tid >> 3, col = (tid & 7) << 4;
          const uint4* g = (const uint4*)(Bb + (long long)(k0 + row) * ldb + (n0 + col));
          uint4 u0 = g[0], u1 = g[1];
          uint4* s = (uint4*)(&Bs[0][0] + row * BN + col);
          s[0] = u0; s[1] = u1;
        } else {
          int row = tid >> 3, col = (tid & 7) << 3;
          uint4 u = *(const uint4*)(Bb + (long long)(k0 + row) * ldb + (n0 + col));
          *(uint4*)(&Bs[0][0] + row * BN + col) = u;
        }
      } else {
        // B stored [N,K]: transpose while staging
        if constexpr (BN == 128) {
          int n = tid >> 1, kk = (tid & 1) << 4;
          const uint4* g = (const uint4*)(Bb + (long long)(n0 + n) * ldb + (k0 + kk));
          uint4 u0 = g[0], u1 = g[1];
          unsigned short tmp[16];
          *(uint4*)&tmp[0] = u0; *(uint4*)&tmp[8] = u1;
#pragma unroll
          for (int t = 0; t < 16; ++t) Bs[0][(kk + t) * BN + n] = tmp[t];
        } else {
          int n = tid >> 2, kk = (tid & 3) << 3;
          uint4 u = *(const uint4*)(Bb + (long long)(n0 + n) * ldb + (k0 + kk));
          unsigned short tmp[8];
          *(uint4*)tmp = u;
#pragma unroll
          for (int t = 0; t < 8; ++t) Bs[0][(kk + t) * BN + n] = tmp[t];
        }
      }
      __syncthreads();
      mma_step<BN>(&As[0][0], &Bs[0][0], acc, wr, wc, half, l15, lane);
      __syncthreads();
    }
  }

  // ---- epilogue: alpha, bias, residual, relu, f32 / bf16 stores ----
#pragma unroll
  for (int ti = 0; ti < 2; ++ti) {
#pragma unroll
    for (int tj = 0; tj < TJ; ++tj) {
      int mb = m0 + wr * 32 + ti * 16;
      int ccol = n0 + wc * (BN / 2) + tj * 16 + l15;
      float bv = bias ? bias[ccol] : 0.0f;
#pragma unroll
      for (int v = 0; v < 8; ++v) {
        int crow = mb + half * 8 + v;
        long long idx = coff + (long long)crow * ldc + ccol;
        float val = alpha * acc[ti][tj][v] + bv;
        if (Res)  val += Res[idx];
        if (relu) val = fmaxf(val, 0.0f);
        if (C)    C[idx] = val;
        if (Cbf)  Cbf[idx] = f32_to_bf16(val);
      }
    }
  }
}

// ---------------- host-side gemm launcher ----------------
template <int BN, int TRANSB>
static inline void gemm(hipStream_t st,
    const void* A, const void* B, float* C, void* Cbf,
    const float* bias, const float* Res,
    int M, int N, int K, int lda, int ldb, int ldc, int Z,
    long long aDiv, long long aS1, long long aMod, long long aS2,
    long long bDiv, long long bS1, long long bMod, long long bS2,
    long long cDiv, long long cS1, long long cMod, long long cS2,
    int relu, float alpha) {
  dim3 grid(N / BN, M / 128, Z), block(256);
  gemm_bf16_wmma<BN, TRANSB><<<grid, block, 0, st>>>(
      (const unsigned short*)A, (const unsigned short*)B,
      C, (unsigned short*)Cbf, bias, Res,
      M, N, K, lda, ldb, ldc,
      aDiv, aS1, aMod, aS2, bDiv, bS1, bMod, bS2, cDiv, cS1, cMod, cS2,
      relu, alpha);
}

extern "C" void kernel_launch(void* const* d_in, const int* in_sizes, int n_in,
                              void* d_out, int out_size, void* d_ws, size_t ws_size,
                              hipStream_t stream) {
  (void)in_sizes; (void)n_in; (void)out_size;
  const int*   x       = (const int*)  d_in[0];
  const float* emb     = (const float*)d_in[1];
  const float* pos_emb = (const float*)d_in[2];
  const float* wq      = (const float*)d_in[3];
  const float* wk      = (const float*)d_in[4];
  const float* wv      = (const float*)d_in[5];
  const float* wo      = (const float*)d_in[6];
  const float* w_in    = (const float*)d_in[7];
  const float* b_in    = (const float*)d_in[8];
  const float* w_out   = (const float*)d_in[9];
  const float* b_out   = (const float*)d_in[10];
  const float* w_unemb = (const float*)d_in[11];
  const float* b_unemb = (const float*)d_in[12];
  float* out = (float*)d_out;

  char* p = (char*)d_ws;
  auto alloc = [&](size_t bytes) -> char* {
    char* r = p; p += (bytes + 255) & ~(size_t)255; return r;
  };
  const size_t nH   = (size_t)BATCH * SEQL * DMODEL;
  const size_t nQKV = (size_t)BATCH * NHEAD * SEQL * DHEAD;
  const size_t nScB = (size_t)NHEAD * SEQL * SEQL;
  const size_t nM1  = (size_t)BATCH * SEQL * DMLP;

  float*          h       = (float*)         alloc(nH * 4);
  unsigned short* hbf     = (unsigned short*)alloc(nH * 2);
  unsigned short* Qbf     = (unsigned short*)alloc(nQKV * 2);
  unsigned short* Kbf     = (unsigned short*)alloc(nQKV * 2);
  unsigned short* Vbf     = (unsigned short*)alloc(nQKV * 2);
  float*          scores  = (float*)         alloc(nScB * 4);
  unsigned short* Obf     = (unsigned short*)alloc(nQKV * 2);
  float*          hattn   = (float*)         alloc(nH * 4);
  unsigned short* hattnbf = (unsigned short*)alloc(nH * 2);
  unsigned short* m1bf    = (unsigned short*)alloc(nM1 * 2);
  unsigned short* hfbf    = (unsigned short*)alloc(nH * 2);
  unsigned short* wqb     = (unsigned short*)alloc((size_t)NHEAD * DMODEL * DHEAD * 2);
  unsigned short* wkb     = (unsigned short*)alloc((size_t)NHEAD * DMODEL * DHEAD * 2);
  unsigned short* wvb     = (unsigned short*)alloc((size_t)NHEAD * DMODEL * DHEAD * 2);
  unsigned short* wob     = (unsigned short*)alloc((size_t)NHEAD * DHEAD * DHEAD * 2);
  unsigned short* winb    = (unsigned short*)alloc((size_t)DMODEL * DMLP * 2);
  unsigned short* woutb   = (unsigned short*)alloc((size_t)DMLP * DMODEL * 2);
  unsigned short* wunb    = (unsigned short*)alloc((size_t)DMODEL * DVOCAB * 2);
  if ((size_t)(p - (char*)d_ws) > ws_size) return;

  auto cvt = [&](const float* src, unsigned short* dst, size_t n) {
    cvt_bf16_kernel<<<dim3((unsigned)((n + 255) / 256)), dim3(256), 0, stream>>>(src, dst, (int)n);
  };
  cvt(wq, wqb, (size_t)NHEAD * DMODEL * DHEAD);
  cvt(wk, wkb, (size_t)NHEAD * DMODEL * DHEAD);
  cvt(wv, wvb, (size_t)NHEAD * DMODEL * DHEAD);
  cvt(wo, wob, (size_t)NHEAD * DHEAD * DHEAD);
  cvt(w_in, winb, (size_t)DMODEL * DMLP);
  cvt(w_out, woutb, (size_t)DMLP * DMODEL);
  cvt(w_unemb, wunb, (size_t)DMODEL * DVOCAB);

  embed_kernel<<<dim3((unsigned)(nH / 256)), dim3(256), 0, stream>>>(x, emb, pos_emb, h, hbf);

  const long long SD  = (long long)SEQL * DMODEL;
  const long long SDh = (long long)SEQL * DHEAD;
  const long long DDh = (long long)DMODEL * DHEAD;

  // QKV projections: z = (b,h)
  gemm<64, 0>(stream, hbf, wqb, nullptr, Qbf, nullptr, nullptr,
      SEQL, DHEAD, DMODEL, DMODEL, DHEAD, DHEAD, BATCH * NHEAD,
      NHEAD, SD, 1, 0,  1, 0, NHEAD, DDh,  1, SDh, 1, 0,  0, 1.0f);
  gemm<64, 0>(stream, hbf, wkb, nullptr, Kbf, nullptr, nullptr,
      SEQL, DHEAD, DMODEL, DMODEL, DHEAD, DHEAD, BATCH * NHEAD,
      NHEAD, SD, 1, 0,  1, 0, NHEAD, DDh,  1, SDh, 1, 0,  0, 1.0f);
  gemm<64, 0>(stream, hbf, wvb, nullptr, Vbf, nullptr, nullptr,
      SEQL, DHEAD, DMODEL, DMODEL, DHEAD, DHEAD, BATCH * NHEAD,
      NHEAD, SD, 1, 0,  1, 0, NHEAD, DDh,  1, SDh, 1, 0,  0, 1.0f);

  // attention (looped over batch to bound the score buffer)
  for (int b = 0; b < BATCH; ++b) {
    const unsigned short* Qb = Qbf + (size_t)b * NHEAD * SEQL * DHEAD;
    const unsigned short* Kb = Kbf + (size_t)b * NHEAD * SEQL * DHEAD;
    const unsigned short* Vb = Vbf + (size_t)b * NHEAD * SEQL * DHEAD;
    unsigned short*       Ob = Obf + (size_t)b * NHEAD * SEQL * DHEAD;

    gemm<128, 1>(stream, Qb, Kb, scores, nullptr, nullptr, nullptr,
        SEQL, SEQL, DHEAD, DHEAD, DHEAD, SEQL, NHEAD,
        1, SDh, 1, 0,  1, SDh, 1, 0,  1, (long long)SEQL * SEQL, 1, 0,  0, 0.125f);

    softmax_causal_kernel<<<dim3(NHEAD * SEQL), dim3(256), 0, stream>>>(scores);

    gemm<64, 0>(stream, (const unsigned short*)scores, Vb, nullptr, Ob, nullptr, nullptr,
        SEQL, DHEAD, SEQL, 2 * SEQL, DHEAD, DHEAD, NHEAD,
        1, (long long)SEQL * 2 * SEQL, 1, 0,  1, SDh, 1, 0,  1, SDh, 1, 0,  0, 1.0f);
  }

  // per-head output projection + residual (heads concat into columns)
  gemm<64, 0>(stream, Obf, wob, hattn, hattnbf, nullptr, h,
      SEQL, DHEAD, DHEAD, DHEAD, DHEAD, DMODEL, BATCH * NHEAD,
      1, SDh, 1, 0,  1, 0, NHEAD, (long long)DHEAD * DHEAD,
      NHEAD, SD, NHEAD, DHEAD,  0, 1.0f);

  // MLP
  gemm<128, 0>(stream, hattnbf, winb, nullptr, m1bf, b_in, nullptr,
      BATCH * SEQL, DMLP, DMODEL, DMODEL, DMLP, DMLP, 1,
      1, 0, 1, 0,  1, 0, 1, 0,  1, 0, 1, 0,  1 /*relu*/, 1.0f);
  gemm<128, 0>(stream, m1bf, woutb, nullptr, hfbf, b_out, hattn,
      BATCH * SEQL, DMODEL, DMLP, DMLP, DMODEL, DMODEL, 1,
      1, 0, 1, 0,  1, 0, 1, 0,  1, 0, 1, 0,  0, 1.0f);

  // unembed
  gemm<128, 0>(stream, hfbf, wunb, out, nullptr, b_unemb, nullptr,
      BATCH * SEQL, DVOCAB, DMODEL, DMODEL, DVOCAB, DVOCAB, 1,
      1, 0, 1, 0,  1, 0, 1, 0,  1, 0, 1, 0,  0, 1.0f);
}
